// SinkhornKnopp_68925635166288
// MI455X (gfx1250) — matrix operand order repaired and checked by reading, not compiled, not run on hardware
//
#include <hip/hip_runtime.h>
#include <math.h>

// ---------------- problem constants (from reference) ----------------
#define M_TOTAL 65536       // B*N = 16*4096
#define DDIM    768
#define KCL     64
#define TEMP_INV 10.0f      // 1/TEMPERATURE
#define NUM_ITERS 3

// ---------------- CDNA5 WMMA types ----------------
typedef __attribute__((ext_vector_type(16))) __bf16 v16bf;
typedef __attribute__((ext_vector_type(8)))  float  v8f;

union Frag { v16bf v; uint4 q[2]; __bf16 e[16]; };

// pointer types for the CDNA5 async global->LDS builtin (typed as int4 pointers)
typedef int v4i __attribute__((vector_size(16)));
typedef __attribute__((address_space(1))) v4i* as1_v4i;
typedef __attribute__((address_space(3))) v4i* as3_v4i;

// optional CDNA5 async global->LDS path (ASYNCcnt-tracked), with safe fallback
#if defined(__has_builtin)
#if __has_builtin(__builtin_amdgcn_global_load_async_to_lds_b128)
#define HAVE_ASYNC_LDS 1
#endif
#if __has_builtin(__builtin_amdgcn_s_wait_asynccnt)
#define HAVE_WAIT_ASYNC 1
#endif
#endif

__device__ __forceinline__ __bf16 f2bf(float f) {
  // round-to-nearest-even f32 -> bf16
  unsigned u = __float_as_uint(f);
  unsigned r = (u + 0x7FFFu + ((u >> 16) & 1u)) >> 16;
  unsigned short s = (unsigned short)r;
  __bf16 out;
  __builtin_memcpy(&out, &s, 2);
  return out;
}

__device__ __forceinline__ float block_reduce_sum_256(float x, float* sm) {
  int t = threadIdx.x;
  sm[t] = x;
  __syncthreads();
  for (int s = 128; s > 0; s >>= 1) {
    if (t < s) sm[t] += sm[t + s];
    __syncthreads();
  }
  float r = sm[0];
  __syncthreads();
  return r;
}

// ---------------- small utility kernels ----------------
__global__ void k_zero(float* p, int n) {
  int i = blockIdx.x * blockDim.x + threadIdx.x;
  if (i < n) p[i] = 0.0f;
}

// normalize 64 centers -> f32 copy + bf16 copy (B-operand of GEMM1)
__global__ __launch_bounds__(256) void k_center_norm(const float* __restrict__ c,
                                                     float* __restrict__ cn,
                                                     __bf16* __restrict__ cbf) {
  __shared__ float sm[256];
  const int k = blockIdx.x;
  const int t = threadIdx.x;
  const float* cr = c + (size_t)k * DDIM;
  float x[3];
  float ss = 0.f;
#pragma unroll
  for (int j = 0; j < 3; ++j) { x[j] = cr[t + j * 256]; ss += x[j] * x[j]; }
  float tot = block_reduce_sum_256(ss, sm);
  float inv = 1.0f / fmaxf(sqrtf(tot), 1e-12f);
#pragma unroll
  for (int j = 0; j < 3; ++j) {
    float nv = x[j] * inv;
    cn[(size_t)k * DDIM + t + j * 256] = nv;
    cbf[(size_t)k * DDIM + t + j * 256] = f2bf(nv);
  }
}

// normalize each feature row and emit bf16 copy; also init v[m]=1
__global__ __launch_bounds__(256) void k_featconv(const float* __restrict__ feat,
                                                  __bf16* __restrict__ fbf,
                                                  float* __restrict__ v) {
  __shared__ float sm[256];
  const int m = blockIdx.x;
  const int t = threadIdx.x;
  const float* fr = feat + (size_t)m * DDIM;
  float x[3];
  float ss = 0.f;
#pragma unroll
  for (int j = 0; j < 3; ++j) { x[j] = fr[t + j * 256]; ss += x[j] * x[j]; }
  float tot = block_reduce_sum_256(ss, sm);
  float inv = 1.0f / fmaxf(sqrtf(tot), 1e-12f);
#pragma unroll
  for (int j = 0; j < 3; ++j)
    fbf[(size_t)m * DDIM + t + j * 256] = f2bf(x[j] * inv);
  if (t == 0) v[m] = 1.0f;
}

// ---------------- GEMM1: E[m,k] = exp(10 * <f_m, c_k>), plus total sum S ----------------
// B operand (96KB bf16 centers) staged once per workgroup into LDS (320KB/WGP on CDNA5).
// A operand streams as pre-converted bf16 (pure b128 loads, no conversion VALU).
// one wave computes a 16(m) x 64(k) tile; 8 waves/block, 512 blocks = 4096 m-tiles.
__global__ __launch_bounds__(256) void k_gemm1(const __bf16* __restrict__ fbf,
                                               const __bf16* __restrict__ cbf,
                                               float* __restrict__ E,
                                               float* __restrict__ S) {
  extern __shared__ __bf16 sB[];   // KCL*DDIM bf16 = 96 KB dynamic LDS

  // ---- cooperative stage of centers into LDS (async, ASYNCcnt-tracked) ----
  {
    const int t = threadIdx.x;
    const uint4* gsrc = (const uint4*)(const void*)cbf;     // 6144 x 16B
    uint4* ldst = (uint4*)(void*)sB;
#if defined(HAVE_ASYNC_LDS)
#pragma unroll
    for (int i = 0; i < 24; ++i) {
      const int c = t + i * 256;
      __builtin_amdgcn_global_load_async_to_lds_b128(
          (as1_v4i)(const void*)(gsrc + c),
          (as3_v4i)(void*)(ldst + c),
          0, 0);
    }
#if defined(HAVE_WAIT_ASYNC)
    __builtin_amdgcn_s_wait_asynccnt(0);
#else
    asm volatile("s_wait_asynccnt 0" ::: "memory");
#endif
#else
#pragma unroll
    for (int i = 0; i < 24; ++i) {
      const int c = t + i * 256;
      ldst[c] = gsrc[c];
    }
#endif
  }
  __syncthreads();

  const int lane = threadIdx.x & 31;
  const int wave = threadIdx.x >> 5;
  const int mtile = blockIdx.x * 8 + wave;
  const int row  = lane & 15;
  const int half = lane >> 4;
  const int kb   = half * 8;          // A-fragment K interleave base (halfs)
  const int m = mtile * 16 + row;
  const __bf16* frb = fbf + (size_t)m * DDIM;

  v8f zacc = {};
  v8f acc[4];
#pragma unroll
  for (int t = 0; t < 4; ++t) acc[t] = zacc;

  const __bf16* bpl[4];
#pragma unroll
  for (int t = 0; t < 4; ++t)
    bpl[t] = sB + (size_t)(t * 16 + row) * DDIM + half * 8;  // B: lane=col, K interleaved like A

  // software pipeline the streaming A operand (two 16B chunks per k-step)
  uint4 a0 = *(const uint4*)(const void*)(frb + kb);
  uint4 a1 = *(const uint4*)(const void*)(frb + kb + 16);

  for (int kc = 0; kc < DDIM / 32; ++kc) {
    Frag a;
    a.q[0] = a0;
    a.q[1] = a1;

    if (kc < DDIM / 32 - 1) {                   // issue next A loads early
      const int d1 = (kc + 1) * 32 + kb;
      __builtin_prefetch(frb + d1 + 32, 0, 1);  // -> global_prefetch_b8
      a0 = *(const uint4*)(const void*)(frb + d1);
      a1 = *(const uint4*)(const void*)(frb + d1 + 16);
    }

    Frag b[4];
#pragma unroll
    for (int t = 0; t < 4; ++t) {
      b[t].q[0] = *(const uint4*)(const void*)(bpl[t] + kc * 32);
      b[t].q[1] = *(const uint4*)(const void*)(bpl[t] + kc * 32 + 16);
    }
#pragma unroll
    for (int t = 0; t < 4; ++t)
      acc[t] = __builtin_amdgcn_wmma_f32_16x16x32_bf16(
          false, a.v, false, b[t].v, (short)0, acc[t], false, false);
  }

  // epilogue: exp, store E (m-major), accumulate global sum
  float lsum = 0.f;
  const int colk = lane & 15;
#pragma unroll
  for (int i = 0; i < 8; ++i) {
    const int mm = mtile * 16 + i + 8 * half;
    float* er = E + (size_t)mm * KCL + colk;
#pragma unroll
    for (int t = 0; t < 4; ++t) {
      float e = __expf(acc[t][i] * TEMP_INV);
      er[t * 16] = e;
      lsum += e;
    }
  }
#pragma unroll
  for (int off = 16; off > 0; off >>= 1) lsum += __shfl_xor(lsum, off, 32);
  if (lane == 0) atomicAdd(S, lsum);
}

// ---------------- Sinkhorn scaling-factor kernels ----------------
__global__ void k_init_u(const float* __restrict__ S, float* __restrict__ u) {
  u[threadIdx.x] = 1.0f / fmaxf(S[0], 1e-12f);   // Q0 = E / sum(E)
}

// r[k] = sum_m W[m,k] * (use_v ? v[m] : 1)   (grid 64 x 256)
__global__ __launch_bounds__(256) void k_rowsum(const float* __restrict__ W,
                                                const float* __restrict__ v,
                                                float* __restrict__ r, int use_v) {
  __shared__ float sm[256];
  const int t = threadIdx.x;
  const int k = t & 63;
  const int sub = t >> 6;
  float acc = 0.f;
  for (int m = blockIdx.x * 4 + sub; m < M_TOTAL; m += 256) {
    float w = W[(size_t)m * KCL + k];
    acc += use_v ? w * v[m] : w;
  }
  sm[t] = acc;
  __syncthreads();
  if (sub == 0) {
    float tot = sm[k] + sm[64 + k] + sm[128 + k] + sm[192 + k];
    atomicAdd(&r[k], tot);
  }
}

__global__ void k_upd_u(float* __restrict__ u, const float* __restrict__ r) {
  int k = threadIdx.x;
  float uk = u[k];
  u[k] = uk / (fmaxf(uk * r[k], 1e-12f) * (float)KCL);
}

// per-m column sum and v update (grid 256 x 256)
__global__ __launch_bounds__(256) void k_colsum(const float* __restrict__ E,
                                                const float* __restrict__ u,
                                                float* __restrict__ v) {
  __shared__ float us[64];
  if (threadIdx.x < 64) us[threadIdx.x] = u[threadIdx.x];
  __syncthreads();
  const int m = blockIdx.x * blockDim.x + threadIdx.x;
  const float* row = E + (size_t)m * KCL;
  float c = 0.f;
#pragma unroll 8
  for (int k = 0; k < KCL; ++k) c += row[k] * us[k];
  float vv = v[m];
  v[m] = vv / (fmaxf(vv * c, 1e-12f) * (float)M_TOTAL);
}

// assignments[m,k] = E[m,k]*u[k]*v[m]*M  (output 1) + bf16 copy for GEMM2
__global__ __launch_bounds__(256) void k_assign(const float* __restrict__ E,
                                                const float* __restrict__ u,
                                                const float* __restrict__ v,
                                                float* __restrict__ out,
                                                __bf16* __restrict__ abf) {
  __shared__ float us[64];
  if (threadIdx.x < 64) us[threadIdx.x] = u[threadIdx.x];
  __syncthreads();
  size_t idx = (size_t)blockIdx.x * blockDim.x + threadIdx.x;
  int m = (int)(idx >> 6);
  int k = (int)(idx & 63);
  float q = E[idx] * us[k] * v[m] * (float)M_TOTAL;
  out[idx] = q;
  abf[idx] = f2bf(q);
}

// ---------------- GEMM2: weighted[k,d] += sum_m assign[m,k] * fnorm[m,d] ----------------
// grid (48 d-tiles, 4 k-tiles, 64 m-chunks of 1024), 1 wave/block, atomic f32 epilogue.
// both operands pre-converted bf16 -> pure u16 loads, no conversion VALU.
__global__ __launch_bounds__(32) void k_gemm2(const __bf16* __restrict__ abf,
                                              const __bf16* __restrict__ fbf,
                                              float* __restrict__ weighted) {
  const int lane  = threadIdx.x;
  const int dtile = blockIdx.x;
  const int ktile = blockIdx.y;
  const int mc    = blockIdx.z;
  const int col  = lane & 15;
  const int half = lane >> 4;
  const int kb = half * 8;    // A K(m)-interleave base
  const int bb = half * 16;   // unused; B uses the kb interleave, like A
  const int krow = ktile * 16 + col;
  const int dcol = dtile * 16 + col;
  v8f acc = {};
  for (int s = 0; s < 32; ++s) {
    const int mbase = mc * 1024 + s * 32;
    Frag a, b;
#pragma unroll
    for (int h = 0; h < 16; ++h) {
      const int mo = (h < 8) ? (kb + h) : (kb + 8 + h);   // = kb+16+(h-8)
      a.e[h] = abf[(size_t)(mbase + mo) * KCL + krow];
    }
#pragma unroll
    for (int h = 0; h < 16; ++h) {
      const int mm = mbase + ((h < 8) ? (kb + h) : (kb + 8 + h));
      b.e[h] = fbf[(size_t)mm * DDIM + dcol];
    }
    acc = __builtin_amdgcn_wmma_f32_16x16x32_bf16(
        false, a.v, false, b.v, (short)0, acc, false, false);
  }
#pragma unroll
  for (int i = 0; i < 8; ++i) {
    const int k = ktile * 16 + i + 8 * half;
    atomicAdd(&weighted[(size_t)k * DDIM + dcol], acc[i]);
  }
}

// ---------------- final EMA center update (output 2) ----------------
__global__ __launch_bounds__(256) void k_final(const float* __restrict__ weighted,
                                               const float* __restrict__ mass,
                                               const float* __restrict__ cn,
                                               float* __restrict__ outc) {
  __shared__ float sm[256];
  const int k = blockIdx.x;
  const int t = threadIdx.x;
  const float ms = fmaxf(mass[k], 1e-6f);
  float nc[3];
  float ss = 0.f;
#pragma unroll
  for (int j = 0; j < 3; ++j) {
    nc[j] = weighted[(size_t)k * DDIM + t + j * 256] / ms;
    ss += nc[j] * nc[j];
  }
  float inv1 = 1.0f / fmaxf(sqrtf(block_reduce_sum_256(ss, sm)), 1e-12f);
  float upd[3];
  float ss2 = 0.f;
#pragma unroll
  for (int j = 0; j < 3; ++j) {
    upd[j] = 0.99f * cn[(size_t)k * DDIM + t + j * 256] + 0.01f * nc[j] * inv1;
    ss2 += upd[j] * upd[j];
  }
  float inv2 = 1.0f / fmaxf(sqrtf(block_reduce_sum_256(ss2, sm)), 1e-12f);
#pragma unroll
  for (int j = 0; j < 3; ++j)
    outc[(size_t)k * DDIM + t + j * 256] = upd[j] * inv2;
}

// ---------------- host launcher ----------------
extern "C" void kernel_launch(void* const* d_in, const int* in_sizes, int n_in,
                              void* d_out, int out_size, void* d_ws, size_t ws_size,
                              hipStream_t stream) {
  const float* feat    = (const float*)d_in[0];  // [16,4096,768]
  const float* centers = (const float*)d_in[1];  // [64,768]
  float* out = (float*)d_out;                    // assignments [M,64] then centers [64,768]

  // workspace partition (256B aligned blocks)
  char* w = (char*)d_ws;
  auto alloc = [&](size_t bytes) -> void* {
    void* p = (void*)w;
    w += (bytes + 255) & ~(size_t)255;
    return p;
  };
  float*  E        = (float*) alloc((size_t)M_TOTAL * KCL * 4);    // 16 MB, L2-resident
  __bf16* fbf      = (__bf16*)alloc((size_t)M_TOTAL * DDIM * 2);   // 96 MB normalized feats bf16
  __bf16* abf      = (__bf16*)alloc((size_t)M_TOTAL * KCL * 2);    // 8 MB assignments bf16
  float*  v        = (float*) alloc((size_t)M_TOTAL * 4);
  float*  cn       = (float*) alloc((size_t)KCL * DDIM * 4);
  __bf16* cbf      = (__bf16*)alloc((size_t)KCL * DDIM * 2);
  float*  weighted = (float*) alloc((size_t)KCL * DDIM * 4);       // contiguous with mass
  float*  mass     = (float*) alloc((size_t)KCL * 4);
  float*  r        = (float*) alloc((size_t)KCL * 4);
  float*  u        = (float*) alloc((size_t)KCL * 4);
  float*  S        = (float*) alloc(4);

  // 1) normalize centers + features (features also converted to bf16 once)
  k_center_norm<<<KCL, 256, 0, stream>>>(centers, cn, cbf);
  k_featconv<<<M_TOTAL, 256, 0, stream>>>(feat, fbf, v);

  // 2) E = exp(logits/T), S = sum(E); centers staged to LDS (96KB dynamic)
  k_zero<<<1, 64, 0, stream>>>(S, 1);
  k_gemm1<<<M_TOTAL / 16 / 8, 256, (size_t)KCL * DDIM * 2, stream>>>(fbf, cbf, E, S);
  k_init_u<<<1, KCL, 0, stream>>>(S, u);

  // 3) Sinkhorn iterations in scaling-factor form (Q = diag(u) E^T diag(v))
  for (int it = 0; it < NUM_ITERS; ++it) {
    k_zero<<<1, KCL, 0, stream>>>(r, KCL);
    k_rowsum<<<64, 256, 0, stream>>>(E, v, r, 1);
    k_upd_u<<<1, KCL, 0, stream>>>(u, r);
    k_colsum<<<M_TOTAL / 256, 256, 0, stream>>>(E, u, v);
  }

  // 4) assignments -> d_out (+ bf16 copy for GEMM2)
  k_assign<<<(M_TOTAL * KCL) / 256, 256, 0, stream>>>(E, u, v, out, abf);

  // 5) EMA center update: mass + weighted (GEMM2) + final normalize
  k_zero<<<(KCL * DDIM + KCL + 255) / 256, 256, 0, stream>>>(weighted, KCL * DDIM + KCL);
  k_rowsum<<<64, 256, 0, stream>>>(out, v, mass, 0);
  k_gemm2<<<dim3(DDIM / 16, KCL / 16, 64), 32, 0, stream>>>(abf, fbf, weighted);
  k_final<<<KCL, 256, 0, stream>>>(weighted, mass, cn, out + (size_t)M_TOTAL * KCL);
}
